// DeformableConv_89876485636819
// MI455X (gfx1250) — matrix operand order repaired and checked by reading, not compiled, not run on hardware
//
#include <hip/hip_runtime.h>
#include <hip/hip_bf16.h>

#define B_    8
#define CIN   128
#define COUT  128
#define HH    112
#define WW    112
#define PIX   (HH*WW)        // 12544
#define KPOS  9
#define KDIM  (CIN*KPOS)     // 1152
#define NKC   (KDIM/32)      // 36 K-chunks of 32

typedef __attribute__((ext_vector_type(16))) __bf16   v16bf;
typedef __attribute__((ext_vector_type(8)))  float    v8f;
typedef __attribute__((ext_vector_type(4)))  unsigned uv4;

union AFrag { v16bf v; uv4 q[2]; };
union BFrag { v16bf v; unsigned u[8]; };

__device__ __forceinline__ unsigned short f2bf(float f) {
  unsigned u = __float_as_uint(f);
  return (unsigned short)((u + 0x7fffu + ((u >> 16) & 1u)) >> 16);
}
// Pack two f32 -> packed bf16 dword. Prefer v_cvt_pk_bf16_f32 when available.
__device__ __forceinline__ unsigned packbf(float f0, float f1) {
#if __has_builtin(__builtin_amdgcn_cvt_pk_bf16_f32)
  auto pk = __builtin_amdgcn_cvt_pk_bf16_f32(f0, f1);
  unsigned r;
  __builtin_memcpy(&r, &pk, sizeof(r));
  return r;
#else
  return (unsigned)f2bf(f0) | ((unsigned)f2bf(f1) << 16);
#endif
}
__device__ __forceinline__ float bflo(unsigned u) { return __uint_as_float(u << 16); }
__device__ __forceinline__ float bfhi(unsigned u) { return __uint_as_float(u & 0xffff0000u); }

// ---------------------------------------------------------------------------
// Kernel 0a: x [B,C,H,W] f32 -> xt [B,H,W,C] bf16  (LDS-tiled transpose)
// ---------------------------------------------------------------------------
__global__ __launch_bounds__(256) void k_transpose(const float* __restrict__ x,
                                                   unsigned short* __restrict__ xt) {
  __shared__ float tile[128][29];
  int bid = blockIdx.x;
  int xt4 = bid & 3;             // 4 x-tiles of 28
  int y   = (bid >> 2) % HH;
  int b   = bid / (HH * 4);
  int x0  = xt4 * 28;
  int tid = threadIdx.x;
  for (int idx = tid; idx < 128 * 28; idx += 256) {
    int c = idx / 28, xx = idx % 28;
    tile[c][xx] = x[(((size_t)b * CIN + c) * HH + y) * WW + x0 + xx];
  }
  __syncthreads();
  for (int idx = tid; idx < 28 * 64; idx += 256) {
    int xx = idx >> 6, c2 = (idx & 63) * 2;
    *(unsigned*)&xt[(((size_t)b * HH + y) * WW + x0 + xx) * CIN + c2] =
        packbf(tile[c2][xx], tile[c2 + 1][xx]);
  }
}

// ---------------------------------------------------------------------------
// Kernel 0b: pack main conv weights w[Cout][C][3][3] -> bf16 A-fragment order
// K-dim index kg = k*128 + c.  Layout: [kc 36][mt 8][lane 32][8 dwords]
// ---------------------------------------------------------------------------
__global__ __launch_bounds__(256) void k_pack_w(const float* __restrict__ w,
                                                unsigned int* __restrict__ wpack) {
  int t = blockIdx.x * 256 + threadIdx.x;
  if (t >= NKC * 8 * 32) return;
  int lane = t & 31;
  int mt   = (t >> 5) & 7;
  int kc   = t >> 8;
  int m    = mt * 16 + (lane & 15);
  unsigned int* dst = wpack + ((size_t)(kc * 8 + mt) * 32 + lane) * 8;
  #pragma unroll
  for (int v = 0; v < 8; ++v) {
    float f[2];
    #pragma unroll
    for (int h = 0; h < 2; ++h) {
      int kloc = ((lane & 16) ? 16 : 0) + 2 * v + h;
      int kg = kc * 32 + kloc;
      int k  = kg >> 7;
      int c  = kg & 127;
      f[h] = w[((size_t)m * CIN + c) * KPOS + k];
    }
    dst[v] = packbf(f[0], f[1]);
  }
}

// ---------------------------------------------------------------------------
// Kernel 0c: pack offset(18)+mod(9) weights, rows padded to 32 with zeros.
// Layout: [kc 36][mt 2][lane 32][8 dwords]
// ---------------------------------------------------------------------------
__global__ __launch_bounds__(256) void k_pack_ow(const float* __restrict__ offw,
                                                 const float* __restrict__ modw,
                                                 unsigned int* __restrict__ wpack2) {
  int t = blockIdx.x * 256 + threadIdx.x;
  if (t >= NKC * 2 * 32) return;
  int lane = t & 31;
  int mt   = (t >> 5) & 1;
  int kc   = t >> 6;
  int m    = mt * 16 + (lane & 15);
  unsigned int* dst = wpack2 + ((size_t)(kc * 2 + mt) * 32 + lane) * 8;
  #pragma unroll
  for (int v = 0; v < 8; ++v) {
    float f[2];
    #pragma unroll
    for (int h = 0; h < 2; ++h) {
      int kloc = ((lane & 16) ? 16 : 0) + 2 * v + h;
      int kg = kc * 32 + kloc;
      int k  = kg >> 7;
      int c  = kg & 127;
      float val = 0.f;
      if (m < 18)      val = offw[((size_t)m * CIN + c) * KPOS + k];
      else if (m < 27) val = modw[((size_t)(m - 18) * CIN + c) * KPOS + k];
      f[h] = val;
    }
    dst[v] = packbf(f[0], f[1]);
  }
}

// ---------------------------------------------------------------------------
// Kernel 1: offset+mask conv as WMMA GEMM (M=32 padded, K=1152, N=64/block).
// 8 waves = 2 M-frags x 4 N-strips. Sigmoid/bias fused in epilogue.
// ---------------------------------------------------------------------------
__global__ __launch_bounds__(256) void k_offmask_gemm(
    const unsigned short* __restrict__ xt,
    const unsigned int*  __restrict__ wpack2,
    const float* __restrict__ offb,
    const float* __restrict__ modb,
    float* __restrict__ offbuf,
    float* __restrict__ maskbuf) {
  __shared__ unsigned short panel[2][64 * 132];
  const int tid  = threadIdx.x;
  const int lane = tid & 31;
  const int wv   = tid >> 5;
  const int wm2  = wv >> 2;            // 0..1 : M strip
  const int wn2  = wv & 3;             // 0..3 : N strip
  const int b    = blockIdx.x / 196;
  const int pix0 = (blockIdx.x % 196) * 64;

  const int sp = tid >> 2;             // pixel 0..63
  const int sq = tid & 3;              // channel quarter
  const int pix = pix0 + sp;
  const int py = pix / WW, pxc = pix % WW;

  v8f acc = {};

  auto build = [&](int kk, int buf) {
    int ki = kk / 3, kj = kk % 3;
    int yy = py - 1 + ki, xx = pxc - 1 + kj;
    bool valid = (yy >= 0 && yy < HH && xx >= 0 && xx < WW);
    unsigned tmp[16];
    if (valid) {
      const uv4* s = (const uv4*)(xt + (((size_t)b * HH + yy) * WW + xx) * CIN + sq * 32);
      #pragma unroll
      for (int i = 0; i < 4; ++i) ((uv4*)tmp)[i] = s[i];
    } else {
      #pragma unroll
      for (int i = 0; i < 16; ++i) tmp[i] = 0u;
    }
    unsigned* dst = (unsigned*)&panel[buf][sp * 132 + sq * 32];
    #pragma unroll
    for (int i = 0; i < 16; ++i) dst[i] = tmp[i];
  };

  auto compute = [&](int kk, int buf) {
    int kh = (lane & 16) ? 16 : 0;
    int pp = wn2 * 16 + (lane & 15);
    #pragma unroll
    for (int cc = 0; cc < 4; ++cc) {
      int kc = kk * 4 + cc;
      AFrag af;
      const uv4* ap = (const uv4*)wpack2 + ((size_t)(kc * 2 + wm2) * 32 + lane) * 2;
      af.q[0] = ap[0]; af.q[1] = ap[1];
      BFrag bf;
      const unsigned short* bp = &panel[buf][pp * 132 + cc * 32 + kh];
      #pragma unroll
      for (int v = 0; v < 8; ++v) bf.u[v] = *(const unsigned*)(bp + 2 * v);
      acc = __builtin_amdgcn_wmma_f32_16x16x32_bf16(false, af.v, false, bf.v,
                                                    (short)0, acc, false, false);
    }
  };

  build(0, 0);
  __syncthreads();
  for (int kk = 0; kk < 9; ++kk) {
    if (kk + 1 < 9) build(kk + 1, (kk + 1) & 1);
    compute(kk, kk & 1);
    __syncthreads();
  }

  int Mb = wm2 * 16 + ((lane & 16) ? 8 : 0);
  int pixel = pix0 + wn2 * 16 + (lane & 15);
  #pragma unroll
  for (int r = 0; r < 8; ++r) {
    int M = Mb + r;
    float v = acc[r];
    if (M < 18) {
      offbuf[((size_t)b * 18 + M) * PIX + pixel] = v + offb[M];
    } else if (M < 27) {
      float z = v + modb[M - 18];
      maskbuf[((size_t)b * 9 + (M - 18)) * PIX + pixel] = 2.f / (1.f + __expf(-z));
    }
  }
}

// ---------------------------------------------------------------------------
// Kernel 2: fused bilinear sampling + main GEMM.
// Per block: 64 pixels, Cout=128. 8 waves in 4(M)x2(N), each 2x2 WMMA tiles.
// Double-buffered LDS B-panel per tap k, K-chunks of 32 channels.
// ---------------------------------------------------------------------------
__global__ __launch_bounds__(256) void k_deform_gemm(
    const unsigned short* __restrict__ xt,
    const unsigned int*  __restrict__ wpack,
    const float* __restrict__ offbuf,
    const float* __restrict__ maskbuf,
    const float* __restrict__ bias,
    float* __restrict__ out) {
  __shared__ unsigned short panel[2][64 * 132];
  const int tid  = threadIdx.x;
  const int lane = tid & 31;
  const int wv   = tid >> 5;
  const int wm   = wv >> 1;            // 0..3 : 32-row M strip
  const int wn   = wv & 1;             // 0..1 : 32-pixel N strip
  const int b    = blockIdx.x / 196;
  const int pix0 = (blockIdx.x % 196) * 64;

  const int sp = tid >> 2;             // pixel 0..63
  const int sq = tid & 3;              // channel quarter (32 ch each)
  const int pix = pix0 + sp;
  const int py = pix / WW, pxc = pix % WW;

  v8f acc[2][2] = {};

  auto build = [&](int kk, int buf) {
    int ki = kk / 3, kj = kk % 3;
    float dy = offbuf[((size_t)b * 18 + 2 * kk) * PIX + pix];
    float dx = offbuf[((size_t)b * 18 + 2 * kk + 1) * PIX + pix];
    float mk = maskbuf[((size_t)b * 9 + kk) * PIX + pix];
    float ys = (float)(py - 1 + ki) + dy;
    float xs = (float)(pxc - 1 + kj) + dx;
    float y0f = floorf(ys), x0f = floorf(xs);
    int iy0 = (int)y0f, ix0 = (int)x0f;
    int iy1 = iy0 + 1,  ix1 = ix0 + 1;
    float ly = ys - y0f, lx = xs - x0f;
    float vy0 = (iy0 >= 0 && iy0 < HH) ? 1.f : 0.f;
    float vy1 = (iy1 >= 0 && iy1 < HH) ? 1.f : 0.f;
    float vx0 = (ix0 >= 0 && ix0 < WW) ? 1.f : 0.f;
    float vx1 = (ix1 >= 0 && ix1 < WW) ? 1.f : 0.f;
    float w00 = (1.f - ly) * (1.f - lx) * vy0 * vx0 * mk;
    float w01 = (1.f - ly) * lx         * vy0 * vx1 * mk;
    float w10 = ly * (1.f - lx)         * vy1 * vx0 * mk;
    float w11 = ly * lx                 * vy1 * vx1 * mk;
    int cy0 = min(max(iy0, 0), HH - 1), cy1 = min(max(iy1, 0), HH - 1);
    int cx0 = min(max(ix0, 0), WW - 1), cx1 = min(max(ix1, 0), WW - 1);
    const uv4* p00 = (const uv4*)(xt + (((size_t)b * HH + cy0) * WW + cx0) * CIN + sq * 32);
    const uv4* p01 = (const uv4*)(xt + (((size_t)b * HH + cy0) * WW + cx1) * CIN + sq * 32);
    const uv4* p10 = (const uv4*)(xt + (((size_t)b * HH + cy1) * WW + cx0) * CIN + sq * 32);
    const uv4* p11 = (const uv4*)(xt + (((size_t)b * HH + cy1) * WW + cx1) * CIN + sq * 32);
    unsigned* dst = (unsigned*)&panel[buf][sp * 132 + sq * 32];
    #pragma unroll
    for (int i = 0; i < 4; ++i) {      // 8 channels per iteration, b128 gathers
      uv4 a0 = p00[i];
      uv4 a1 = p01[i];
      uv4 a2 = p10[i];
      uv4 a3 = p11[i];
      #pragma unroll
      for (int j = 0; j < 4; ++j) {
        unsigned u00 = a0[j], u01 = a1[j], u10 = a2[j], u11 = a3[j];
        float f0 = w00 * bflo(u00) + w01 * bflo(u01) + w10 * bflo(u10) + w11 * bflo(u11);
        float f1 = w00 * bfhi(u00) + w01 * bfhi(u01) + w10 * bfhi(u10) + w11 * bfhi(u11);
        dst[i * 4 + j] = packbf(f0, f1);
      }
    }
  };

  auto compute = [&](int kk, int buf) {
    int kh = (lane & 16) ? 16 : 0;
    #pragma unroll
    for (int cc = 0; cc < 4; ++cc) {
      int kc = kk * 4 + cc;
      AFrag af[2];
      #pragma unroll
      for (int mi = 0; mi < 2; ++mi) {
        int mt = wm * 2 + mi;
        const uv4* ap = (const uv4*)wpack + ((size_t)(kc * 8 + mt) * 32 + lane) * 2;
        af[mi].q[0] = ap[0]; af[mi].q[1] = ap[1];
      }
      if (kc + 1 < NKC)
        __builtin_prefetch((const uv4*)wpack + ((size_t)((kc + 1) * 8 + wm * 2) * 32 + lane) * 2, 0, 0);
      BFrag bf[2];
      #pragma unroll
      for (int ni = 0; ni < 2; ++ni) {
        int pp = wn * 32 + ni * 16 + (lane & 15);
        const unsigned short* bp = &panel[buf][pp * 132 + cc * 32 + kh];
        #pragma unroll
        for (int v = 0; v < 8; ++v) bf[ni].u[v] = *(const unsigned*)(bp + 2 * v);
      }
      #pragma unroll
      for (int mi = 0; mi < 2; ++mi)
        #pragma unroll
        for (int ni = 0; ni < 2; ++ni)
          acc[mi][ni] = __builtin_amdgcn_wmma_f32_16x16x32_bf16(
              false, af[mi].v, false, bf[ni].v, (short)0, acc[mi][ni], false, false);
    }
  };

  build(0, 0);
  __syncthreads();
  for (int kk = 0; kk < 9; ++kk) {
    if (kk + 1 < 9) build(kk + 1, (kk + 1) & 1);
    compute(kk, kk & 1);
    __syncthreads();
  }

  #pragma unroll
  for (int mi = 0; mi < 2; ++mi) {
    int Mb = (wm * 2 + mi) * 16 + ((lane & 16) ? 8 : 0);
    #pragma unroll
    for (int ni = 0; ni < 2; ++ni) {
      int pixel = pix0 + wn * 32 + ni * 16 + (lane & 15);
      #pragma unroll
      for (int r = 0; r < 8; ++r) {
        int M = Mb + r;
        out[((size_t)b * COUT + M) * PIX + pixel] = acc[mi][ni][r] + bias[M];
      }
    }
  }
}

// ---------------------------------------------------------------------------
extern "C" void kernel_launch(void* const* d_in, const int* in_sizes, int n_in,
                              void* d_out, int out_size, void* d_ws, size_t ws_size,
                              hipStream_t stream) {
  const float* x    = (const float*)d_in[0];
  const float* offw = (const float*)d_in[1];
  const float* offb = (const float*)d_in[2];
  const float* modw = (const float*)d_in[3];
  const float* modb = (const float*)d_in[4];
  const float* w    = (const float*)d_in[5];
  const float* bias = (const float*)d_in[6];
  float* out = (float*)d_out;

  char* ws = (char*)d_ws;
  unsigned short* xt = (unsigned short*)ws;                     // 25,690,112 B
  size_t o = (size_t)B_ * HH * WW * CIN * 2;
  unsigned int* wpack = (unsigned int*)(ws + o);                //    294,912 B
  o += (size_t)NKC * 8 * 32 * 32;
  unsigned int* wpack2 = (unsigned int*)(ws + o);               //     73,728 B
  o += (size_t)NKC * 2 * 32 * 32;
  float* offbuf = (float*)(ws + o);                             //  7,225,344 B
  o += (size_t)B_ * 18 * PIX * 4;
  float* maskbuf = (float*)(ws + o);                            //  3,612,672 B

  k_transpose  <<<B_ * HH * 4, 256, 0, stream>>>(x, xt);
  k_pack_w     <<<(NKC * 8 * 32 + 255) / 256, 256, 0, stream>>>(w, wpack);
  k_pack_ow    <<<(NKC * 2 * 32 + 255) / 256, 256, 0, stream>>>(offw, modw, wpack2);
  k_offmask_gemm<<<B_ * 196, 256, 0, stream>>>(xt, wpack2, offb, modb, offbuf, maskbuf);
  k_deform_gemm <<<B_ * 196, 256, 0, stream>>>(xt, wpack, offbuf, maskbuf, bias, out);
}
